// MineBlock_80092550135894
// MI455X (gfx1250) — matrix operand order repaired and checked by reading, not compiled, not run on hardware
//
#include <hip/hip_runtime.h>

// ---------------- problem constants ----------------
#define B_   8
#define C_   128
#define H_   64
#define W_   64
#define L_   4096            // H_*W_
#define S_   8
#define R_   8
#define NTOK 32768           // B_*L_
#define EPS_ 1e-5f

typedef __attribute__((ext_vector_type(16))) __bf16 v16bf;
typedef __attribute__((ext_vector_type(8)))  __bf16 v8bf;
typedef __attribute__((ext_vector_type(8)))  float  v8f;

// fp32 -> bf16 round-to-nearest-even (used only in producer/convert kernels)
__device__ __forceinline__ __bf16 f2bf(float x) {
    unsigned u = __builtin_bit_cast(unsigned, x);
    unsigned r = (u + 0x7FFFu + ((u >> 16) & 1u)) >> 16;
    unsigned short h = (unsigned short)r;
    return __builtin_bit_cast(__bf16, h);
}

__device__ __forceinline__ float siluf(float x) { return x / (1.f + __expf(-x)); }
__device__ __forceinline__ float softplusf(float x) {
    return (x > 20.f) ? x : log1pf(__expf(x));
}

// ---------------- fragment loaders: raw packed bf16, no conversion ----------------
// A fragment (16x32 bf16): lane holds row m0+(l&15); elems 0..7 = K kb+0..7,
// elems 8..15 = K 16+kb+0..7 (kb = 8*(l>>4)) -> two 16B chunks 32 elems... 16 elems apart.
__device__ __forceinline__ v16bf ldfragA(const __bf16* p) {
    v8bf lo = *(const v8bf*)p;
    v8bf hi = *(const v8bf*)(p + 16);
    return __builtin_shufflevector(lo, hi, 0,1,2,3,4,5,6,7,8,9,10,11,12,13,14,15);
}
// B fragment (32x16 bf16): lane holds col n0+(l&15); elems 0..7 = K kb+0..7, elems 8..15 = K 16+kb+0..7 (kb = 8*(l>>4)).
__device__ __forceinline__ v16bf ldfragB(const __bf16* p) {
    v8bf lo = *(const v8bf*)p;
    v8bf hi = *(const v8bf*)(p + 16);
    return __builtin_shufflevector(lo, hi, 0,1,2,3,4,5,6,7,8,9,10,11,12,13,14,15);
}

// ---------------- WMMA GEMM: out[M,N] (fp32) = A[M,K] @ W[N,K]^T (bf16 in) ----------
// block (32,4): 4 waves, each wave -> 32x32 tile; block tile 128x32.
// Requires M%128==0, N%32==0, K==128 (true for all uses here).
__global__ __launch_bounds__(128) void k_gemm_wmma(const __bf16* __restrict__ A,
                                                   const __bf16* __restrict__ W,
                                                   float* __restrict__ out,
                                                   int M, int N, int K) {
    const int lane = threadIdx.x;                 // 0..31 (wave32)
    const int wave = threadIdx.y;                 // 0..3
    const int m0 = blockIdx.y * 128 + wave * 32;
    const int n0 = blockIdx.x * 32;
    const int r  = lane & 15;
    const int kbA = (lane >> 4) << 3;             // 0 or 8
    const int kbB = (lane >> 4) << 3;             // 0 or 8
    const __bf16* Ar0 = A + (size_t)(m0 + r)      * K + kbA;
    const __bf16* Ar1 = A + (size_t)(m0 + 16 + r) * K + kbA;
    const __bf16* Wr0 = W + (size_t)(n0 + r)      * K + kbB;
    const __bf16* Wr1 = W + (size_t)(n0 + 16 + r) * K + kbB;
    v8f c00 = {}, c01 = {}, c10 = {}, c11 = {};
#pragma unroll
    for (int k = 0; k < 128; k += 32) {           // K==128 for every call site
        __builtin_prefetch(Ar0 + k + 128, 0, 1);  // gfx1250 global_prefetch path
        v16bf a0 = ldfragA(Ar0 + k);
        v16bf a1 = ldfragA(Ar1 + k);
        v16bf b0 = ldfragB(Wr0 + k);
        v16bf b1 = ldfragB(Wr1 + k);
        c00 = __builtin_amdgcn_wmma_f32_16x16x32_bf16(false, a0, false, b0, (short)0, c00, false, false);
        c01 = __builtin_amdgcn_wmma_f32_16x16x32_bf16(false, a0, false, b1, (short)0, c01, false, false);
        c10 = __builtin_amdgcn_wmma_f32_16x16x32_bf16(false, a1, false, b0, (short)0, c10, false, false);
        c11 = __builtin_amdgcn_wmma_f32_16x16x32_bf16(false, a1, false, b1, (short)0, c11, false, false);
    }
    const int nn = n0 + (lane & 15);
    const int mb = m0 + ((lane >> 4) << 3);
#pragma unroll
    for (int i = 0; i < 8; ++i) {
        out[(size_t)(mb + i)      * N + nn]      = c00[i];
        out[(size_t)(mb + i)      * N + nn + 16] = c01[i];
        out[(size_t)(mb + 16 + i) * N + nn]      = c10[i];
        out[(size_t)(mb + 16 + i) * N + nn + 16] = c11[i];
    }
}

// ---------------- generic fp32 -> bf16 convert (weights; n % 256 == 0) --------------
__global__ void k_cvt_bf16(const float* __restrict__ in, __bf16* __restrict__ out) {
    int i = blockIdx.x * blockDim.x + threadIdx.x;
    out[i] = f2bf(in[i]);
}

// ---------------- x (B,C,H,W) -> xt (B*L, C) token-major, bf16 ----------------
__global__ void k_transpose_tok(const float* __restrict__ x, __bf16* __restrict__ xt) {
    int idx = blockIdx.x * blockDim.x + threadIdx.x;           // B_*L_*C_
    int c  = idx & (C_ - 1);
    int hw = (idx >> 7) & (L_ - 1);
    int b  = idx >> 19;
    xt[((size_t)b * L_ + hw) * C_ + c] = f2bf(x[((size_t)b * C_ + c) * L_ + hw]);
}

// ---------------- SS2D: depthwise 3x3 conv (pad 1) + SiLU; xi from g_in token-major ----
__global__ void k_ss2d_conv(const float* __restrict__ g_in, const float* __restrict__ cw,
                            const float* __restrict__ cb, float* __restrict__ xc) {
    int idx = blockIdx.x * blockDim.x + threadIdx.x;           // B_*C_*L_
    int hw = idx & (L_ - 1);
    int c  = (idx >> 12) & (C_ - 1);
    int b  = idx >> 19;
    int h = hw >> 6, w = hw & 63;
    float acc = cb[c];
#pragma unroll
    for (int kh = 0; kh < 3; ++kh) {
        int hh = h + kh - 1;
        if (hh < 0 || hh >= H_) continue;
#pragma unroll
        for (int kw = 0; kw < 3; ++kw) {
            int ww = w + kw - 1;
            if (ww < 0 || ww >= W_) continue;
            acc += cw[c * 9 + kh * 3 + kw] *
                   g_in[((size_t)b * L_ + (hh * W_ + ww)) * 256 + c];
        }
    }
    xc[((size_t)b * C_ + c) * L_ + hw] = siluf(acc);
}

// ---------------- build 4-direction xs (B,4,C,L) from xc (B,C,L) ----------------
__global__ void k_build_xs(const float* __restrict__ xc, float* __restrict__ xs) {
    size_t idx = (size_t)blockIdx.x * blockDim.x + threadIdx.x; // B_*4*C_*L_
    int l  = (int)(idx & (L_ - 1));
    size_t rest = idx >> 12;
    int c  = (int)(rest & (C_ - 1));
    int bg = (int)(rest >> 7);
    int g  = bg & 3;
    int b  = bg >> 2;
    int src;
    if (g == 0)      src = l;
    else if (g == 1) src = (l & 63) * 64 + (l >> 6);
    else if (g == 2) src = L_ - 1 - l;
    else { int l2 = L_ - 1 - l; src = (l2 & 63) * 64 + (l2 >> 6); }
    xs[idx] = xc[((size_t)b * C_ + c) * L_ + src];
}

// ---------------- proj[b,g,d,l] = sum_c xs[b,g,c,l] * xproj_w[g,d,c]  (d=24) --------
__global__ void k_ss2d_proj(const float* __restrict__ xs, const float* __restrict__ pw,
                            float* __restrict__ proj) {
    size_t idx = (size_t)blockIdx.x * blockDim.x + threadIdx.x; // B_*4*24*L_
    int l  = (int)(idx & (L_ - 1));
    size_t rest = idx >> 12;
    int d  = (int)(rest % 24);
    int bg = (int)(rest / 24);
    int g  = bg & 3;
    const float* xp = xs + ((size_t)bg * C_) * L_ + l;
    const float* wp = pw + ((size_t)g * 24 + d) * C_;
    float acc = 0.f;
#pragma unroll 8
    for (int c = 0; c < C_; ++c) acc += xp[(size_t)c * L_] * wp[c];
    proj[idx] = acc;
}

// ---------------- dts[b,g,c,l] = softplus(proj[r]·dt_w[g,c,r] + dt_b[g,c]) ----------
__global__ void k_ss2d_dt(const float* __restrict__ proj, const float* __restrict__ dw,
                          const float* __restrict__ db, float* __restrict__ dts) {
    size_t idx = (size_t)blockIdx.x * blockDim.x + threadIdx.x; // B_*4*C_*L_
    int l  = (int)(idx & (L_ - 1));
    size_t rest = idx >> 12;
    int c  = (int)(rest & (C_ - 1));
    int bg = (int)(rest >> 7);
    int g  = bg & 3;
    float acc = db[g * C_ + c];
    const float* pp = proj + ((size_t)bg * 24) * L_ + l;
    const float* wp = dw + ((size_t)g * C_ + c) * R_;
#pragma unroll
    for (int r = 0; r < R_; ++r) acc += pp[(size_t)r * L_] * wp[r];
    dts[idx] = softplusf(acc);
}

// ---------------- SS2D selective scan: lane = state; 4 sequences / wave --------------
__global__ void k_ss2d_scan(const float* __restrict__ xs, const float* __restrict__ dts,
                            const float* __restrict__ proj, const float* __restrict__ Alog,
                            const float* __restrict__ Dp, float* __restrict__ ys) {
    int lane = threadIdx.x & 31;
    int s = lane & 7, q = lane >> 3;
    int waveId = blockIdx.x * (blockDim.x >> 5) + (threadIdx.x >> 5);
    int seq = waveId * 4 + q;                                   // 0..B_*4*C_-1
    int c  = seq & (C_ - 1);
    int bg = seq >> 7;
    int g  = bg & 3;
    float A  = -__expf(Alog[((size_t)g * C_ + c) * S_ + s]);
    float Dv = Dp[g * C_ + c];
    const float* xp  = xs  + ((size_t)bg * C_ + c) * L_;
    const float* dtp = dts + ((size_t)bg * C_ + c) * L_;
    const float* Bp  = proj + ((size_t)bg * 24 + 8 + s)  * L_;
    const float* Cp  = proj + ((size_t)bg * 24 + 16 + s) * L_;
    float* yp = ys + ((size_t)bg * C_ + c) * L_;
    float h = 0.f;
    for (int l = 0; l < L_; ++l) {
        float dt = dtp[l], xv = xp[l];
        h = h * __expf(dt * A) + dt * Bp[l] * xv;
        float part = h * Cp[l];
        part += __shfl_xor(part, 1, 32);
        part += __shfl_xor(part, 2, 32);
        part += __shfl_xor(part, 4, 32);
        if (s == 0) yp[l] = part + Dv * xv;
    }
}

// ------- merge 4 dirs + LayerNorm(C) + SiLU(z) gate -> ytok (N,C) bf16 --------------
__global__ void k_ss2d_merge(const float* __restrict__ ys, const float* __restrict__ g_in,
                             const float* __restrict__ lg, const float* __restrict__ lb,
                             __bf16* __restrict__ ytok) {
    int lane = threadIdx.x & 31;
    int tok  = blockIdx.x * (blockDim.x >> 5) + (threadIdx.x >> 5); // 0..NTOK-1
    int b = tok >> 12, l = tok & (L_ - 1);
    int lT = (l & 63) * 64 + (l >> 6);
    float v[4], sum = 0.f, sq = 0.f;
#pragma unroll
    for (int i = 0; i < 4; ++i) {
        int c = lane + 32 * i;
        float t0 = ys[(((size_t)(b * 4 + 0) * C_) + c) * L_ + l];
        float t2 = ys[(((size_t)(b * 4 + 2) * C_) + c) * L_ + (L_ - 1 - l)];
        float t1 = ys[(((size_t)(b * 4 + 1) * C_) + c) * L_ + lT];
        float t3 = ys[(((size_t)(b * 4 + 3) * C_) + c) * L_ + (L_ - 1 - lT)];
        float y = t0 + t1 + t2 + t3;
        v[i] = y; sum += y; sq += y * y;
    }
#pragma unroll
    for (int m = 1; m < 32; m <<= 1) {
        sum += __shfl_xor(sum, m, 32);
        sq  += __shfl_xor(sq,  m, 32);
    }
    float mean = sum * (1.f / C_);
    float var  = sq * (1.f / C_) - mean * mean;
    float inv  = rsqrtf(var + EPS_);
#pragma unroll
    for (int i = 0; i < 4; ++i) {
        int c = lane + 32 * i;
        float z = g_in[(size_t)tok * 256 + 128 + c];
        ytok[(size_t)tok * C_ + c] = f2bf(((v[i] - mean) * inv * lg[c] + lb[c]) * siluf(z));
    }
}

// ------- reflectance branch: dwconv3x3 on x -> seq token-major bf16 -----------------
__global__ void k_dwconv_x(const float* __restrict__ x, const float* __restrict__ cw,
                           const float* __restrict__ cb, __bf16* __restrict__ seq) {
    int idx = blockIdx.x * blockDim.x + threadIdx.x;           // B_*L_*C_
    int c  = idx & (C_ - 1);
    int hw = (idx >> 7) & (L_ - 1);
    int b  = idx >> 19;
    int h = hw >> 6, w = hw & 63;
    float acc = cb[c];
    const float* xp = x + ((size_t)b * C_ + c) * L_;
#pragma unroll
    for (int kh = 0; kh < 3; ++kh) {
        int hh = h + kh - 1;
        if (hh < 0 || hh >= H_) continue;
#pragma unroll
        for (int kw = 0; kw < 3; ++kw) {
            int ww = w + kw - 1;
            if (ww < 0 || ww >= W_) continue;
            acc += cw[c * 9 + kh * 3 + kw] * xp[hh * W_ + ww];
        }
    }
    seq[((size_t)b * L_ + hw) * C_ + c] = f2bf(acc);
}

// ------- Mamba causal conv1d (k=4, pad left 3) + SiLU -> xc_m (B,C,L) ---------------
__global__ void k_m_conv1d(const float* __restrict__ g_in, const float* __restrict__ cw,
                           const float* __restrict__ cb, float* __restrict__ xcm) {
    int idx = blockIdx.x * blockDim.x + threadIdx.x;           // B_*C_*L_
    int l = idx & (L_ - 1);
    int c = (idx >> 12) & (C_ - 1);
    int b = idx >> 19;
    float acc = cb[c];
#pragma unroll
    for (int j = 0; j < 4; ++j) {
        int lp = l - 3 + j;
        if (lp >= 0)
            acc += cw[c * 4 + j] * g_in[((size_t)b * L_ + lp) * 256 + c];
    }
    xcm[((size_t)b * C_ + c) * L_ + l] = siluf(acc);
}

__global__ void k_m_proj(const float* __restrict__ xcm, const float* __restrict__ pw,
                         float* __restrict__ proj) {
    size_t idx = (size_t)blockIdx.x * blockDim.x + threadIdx.x; // B_*24*L_
    int l = (int)(idx & (L_ - 1));
    size_t rest = idx >> 12;
    int d = (int)(rest % 24);
    int b = (int)(rest / 24);
    const float* xp = xcm + ((size_t)b * C_) * L_ + l;
    const float* wp = pw + (size_t)d * C_;
    float acc = 0.f;
#pragma unroll 8
    for (int c = 0; c < C_; ++c) acc += xp[(size_t)c * L_] * wp[c];
    proj[idx] = acc;
}

__global__ void k_m_dt(const float* __restrict__ proj, const float* __restrict__ dw,
                       const float* __restrict__ db, float* __restrict__ dts) {
    size_t idx = (size_t)blockIdx.x * blockDim.x + threadIdx.x; // B_*C_*L_
    int l = (int)(idx & (L_ - 1));
    size_t rest = idx >> 12;
    int c = (int)(rest & (C_ - 1));
    int b = (int)(rest >> 7);
    float acc = db[c];
    const float* pp = proj + ((size_t)b * 24) * L_ + l;
    const float* wp = dw + (size_t)c * R_;
#pragma unroll
    for (int r = 0; r < R_; ++r) acc += pp[(size_t)r * L_] * wp[r];
    dts[idx] = softplusf(acc);
}

__global__ void k_m_scan(const float* __restrict__ xcm, const float* __restrict__ dts,
                         const float* __restrict__ proj, const float* __restrict__ Alog,
                         const float* __restrict__ Dp, float* __restrict__ ys) {
    int lane = threadIdx.x & 31;
    int s = lane & 7, q = lane >> 3;
    int waveId = blockIdx.x * (blockDim.x >> 5) + (threadIdx.x >> 5);
    int seq = waveId * 4 + q;                                   // 0..B_*C_-1
    int c = seq & (C_ - 1);
    int b = seq >> 7;
    float A  = -__expf(Alog[(size_t)c * S_ + s]);
    float Dv = Dp[c];
    const float* xp  = xcm + ((size_t)b * C_ + c) * L_;
    const float* dtp = dts + ((size_t)b * C_ + c) * L_;
    const float* Bp  = proj + ((size_t)b * 24 + 8 + s)  * L_;
    const float* Cp  = proj + ((size_t)b * 24 + 16 + s) * L_;
    float* yp = ys + ((size_t)b * C_ + c) * L_;
    float h = 0.f;
    for (int l = 0; l < L_; ++l) {
        float dt = dtp[l], xv = xp[l];
        h = h * __expf(dt * A) + dt * Bp[l] * xv;
        float part = h * Cp[l];
        part += __shfl_xor(part, 1, 32);
        part += __shfl_xor(part, 2, 32);
        part += __shfl_xor(part, 4, 32);
        if (s == 0) yp[l] = part + Dv * xv;
    }
}

__global__ void k_m_gate(const float* __restrict__ ys, const float* __restrict__ g_in,
                         __bf16* __restrict__ ytok) {
    int idx = blockIdx.x * blockDim.x + threadIdx.x;           // B_*L_*C_
    int c = idx & (C_ - 1);
    int l = (idx >> 7) & (L_ - 1);
    int b = idx >> 19;
    float z = g_in[((size_t)b * L_ + l) * 256 + 128 + c];
    ytok[((size_t)b * L_ + l) * C_ + c] =
        f2bf(ys[((size_t)b * C_ + c) * L_ + l] * siluf(z));
}

// ---------------- E0 gate + final fusion ----------------
__global__ void k_e0(const float* __restrict__ x, const float* __restrict__ iw,
                     float* __restrict__ e0) {
    int tok = blockIdx.x * blockDim.x + threadIdx.x;           // NTOK
    int b = tok >> 12, l = tok & (L_ - 1);
    float acc = 0.f;
#pragma unroll 8
    for (int c = 0; c < C_; ++c) acc += x[((size_t)b * C_ + c) * L_ + l] * iw[c];
    e0[tok] = 1.f / (1.f + __expf(-acc));
}

__global__ void k_fuse(const float* __restrict__ x, const float* __restrict__ fil,
                       const float* __restrict__ frf, const float* __restrict__ e0,
                       float* __restrict__ out) {
    int idx = blockIdx.x * blockDim.x + threadIdx.x;           // B_*C_*L_
    int hw = idx & (L_ - 1);
    int c  = (idx >> 12) & (C_ - 1);
    int b  = idx >> 19;
    int tok = b * L_ + hw;
    float e = e0[tok];
    size_t tc = (size_t)tok * C_ + c;
    out[idx] = (1.f - e) * fil[tc] + e * frf[tc] + x[idx];
}

// ---------------- host-side orchestration ----------------
extern "C" void kernel_launch(void* const* d_in, const int* in_sizes, int n_in,
                              void* d_out, int out_size, void* d_ws, size_t ws_size,
                              hipStream_t stream) {
    const float* x         = (const float*)d_in[0];
    const float* in_w      = (const float*)d_in[1];
    const float* conv_w    = (const float*)d_in[2];
    const float* conv_b    = (const float*)d_in[3];
    const float* xproj_w   = (const float*)d_in[4];
    const float* dt_w      = (const float*)d_in[5];
    const float* dt_b      = (const float*)d_in[6];
    const float* Alog      = (const float*)d_in[7];
    const float* Dp        = (const float*)d_in[8];
    const float* ln_g      = (const float*)d_in[9];
    const float* ln_b      = (const float*)d_in[10];
    const float* out_w     = (const float*)d_in[11];
    const float* dw_w      = (const float*)d_in[12];
    const float* dw_b      = (const float*)d_in[13];
    const float* m_in_w    = (const float*)d_in[14];
    const float* m_conv_w  = (const float*)d_in[15];
    const float* m_conv_b  = (const float*)d_in[16];
    const float* m_xproj_w = (const float*)d_in[17];
    const float* m_dt_w    = (const float*)d_in[18];
    const float* m_dt_b    = (const float*)d_in[19];
    const float* m_Alog    = (const float*)d_in[20];
    const float* m_D       = (const float*)d_in[21];
    const float* m_out_w   = (const float*)d_in[22];
    const float* illum_w   = (const float*)d_in[23];
    float* out = (float*)d_out;

    // workspace layout (units: floats; bf16 regions reuse float-sized slots)
    float* ws = (float*)d_ws;
    float*  xtf   = ws;                      // 4,194,304 slot: xt_bf / seq_bf (bf16)
    float*  g_in  = xtf   + 4194304;         // 8,388,608
    float*  xc    = g_in  + 8388608;         // 4,194,304  (aliased: xc_m)
    float*  xs    = xc    + 4194304;         // 16,777,216 (aliased after scan)
    float*  proj  = xs    + 16777216;        // 3,145,728
    float*  dts   = proj  + 3145728;         // 16,777,216
    float*  ys    = dts   + 16777216;        // 16,777,216
    float*  ytokf = ys    + 16777216;        // 4,194,304 slot: ytok_bf / ytokm_bf
    float*  fil   = ytokf + 4194304;         // 4,194,304
    float*  ginm  = fil   + 4194304;         // 8,388,608
    float*  frf   = ginm  + 8388608;         // 4,194,304
    float*  e0    = frf   + 4194304;         // 32,768
    float*  wbase = e0    + 32768;           // 49,152 floats -> 98,304 bf16 weights
    // typed views / aliases
    __bf16* xt_bf    = (__bf16*)xtf;         // also seq_bf (sequential reuse)
    __bf16* seq_bf   = (__bf16*)xtf;
    __bf16* ytok_bf  = (__bf16*)ytokf;       // also ytokm_bf (sequential reuse)
    __bf16* ytokm_bf = (__bf16*)ytokf;
    float*  xcm   = xc;
    float*  projm = xs;                      //   786,432
    float*  dtsm  = xs + 1048576;            // 4,194,304
    float*  ysm   = xs + 6291456;            // 4,194,304
    __bf16* inw_bf   = (__bf16*)wbase;       // 32,768
    __bf16* outw_bf  = inw_bf  + 32768;      // 16,384
    __bf16* minw_bf  = outw_bf + 16384;      // 32,768
    __bf16* moutw_bf = minw_bf + 32768;      // 16,384

    const int TB = 256;
    dim3 gblk(32, 4);

    // ---- one-shot weight conversions to bf16 ----
    k_cvt_bf16<<<32768/TB, TB, 0, stream>>>(in_w,    inw_bf);
    k_cvt_bf16<<<16384/TB, TB, 0, stream>>>(out_w,   outw_bf);
    k_cvt_bf16<<<32768/TB, TB, 0, stream>>>(m_in_w,  minw_bf);
    k_cvt_bf16<<<16384/TB, TB, 0, stream>>>(m_out_w, moutw_bf);

    // ---- SS2D (illumination) branch ----
    k_transpose_tok<<<(B_*C_*L_)/TB, TB, 0, stream>>>(x, xt_bf);
    k_gemm_wmma<<<dim3(256/32, NTOK/128), gblk, 0, stream>>>(xt_bf, inw_bf, g_in, NTOK, 256, C_);
    k_ss2d_conv<<<(B_*C_*L_)/TB, TB, 0, stream>>>(g_in, conv_w, conv_b, xc);
    k_build_xs<<<(B_*4*C_*L_)/TB, TB, 0, stream>>>(xc, xs);
    k_ss2d_proj<<<(B_*4*24*L_)/TB, TB, 0, stream>>>(xs, xproj_w, proj);
    k_ss2d_dt<<<(B_*4*C_*L_)/TB, TB, 0, stream>>>(proj, dt_w, dt_b, dts);
    k_ss2d_scan<<<(B_*4*C_*8)/TB, TB, 0, stream>>>(xs, dts, proj, Alog, Dp, ys);
    k_ss2d_merge<<<NTOK/8, TB, 0, stream>>>(ys, g_in, ln_g, ln_b, ytok_bf);
    k_gemm_wmma<<<dim3(128/32, NTOK/128), gblk, 0, stream>>>(ytok_bf, outw_bf, fil, NTOK, C_, C_);

    // ---- Mamba (reflectance) branch ----
    k_dwconv_x<<<(B_*L_*C_)/TB, TB, 0, stream>>>(x, dw_w, dw_b, seq_bf);
    k_gemm_wmma<<<dim3(256/32, NTOK/128), gblk, 0, stream>>>(seq_bf, minw_bf, ginm, NTOK, 256, C_);
    k_m_conv1d<<<(B_*C_*L_)/TB, TB, 0, stream>>>(ginm, m_conv_w, m_conv_b, xcm);
    k_m_proj<<<(B_*24*L_)/TB, TB, 0, stream>>>(xcm, m_xproj_w, projm);
    k_m_dt<<<(B_*C_*L_)/TB, TB, 0, stream>>>(projm, m_dt_w, m_dt_b, dtsm);
    k_m_scan<<<(B_*C_*8)/TB, TB, 0, stream>>>(xcm, dtsm, projm, m_Alog, m_D, ysm);
    k_m_gate<<<(B_*L_*C_)/TB, TB, 0, stream>>>(ysm, ginm, ytokm_bf);
    k_gemm_wmma<<<dim3(128/32, NTOK/128), gblk, 0, stream>>>(ytokm_bf, moutw_bf, frf, NTOK, C_, C_);

    // ---- physics-prior gate + residual fusion ----
    k_e0<<<NTOK/TB, TB, 0, stream>>>(x, illum_w, e0);
    k_fuse<<<(B_*C_*L_)/TB, TB, 0, stream>>>(x, fil, frf, e0, out);

    (void)in_sizes; (void)n_in; (void)out_size; (void)ws_size;
}
